// T2SBlock_19610820674349
// MI455X (gfx1250) — matrix operand order repaired and checked by reading, not compiled, not run on hardware
//
#include <hip/hip_runtime.h>
#include <hip/hip_bf16.h>

// ---------------------------------------------------------------------------
// MI455X (gfx1250) transformer block: bf16 WMMA GEMMs + fp32 softmax/LN.
// B=2, S=2048, H=1024, nh=16, hd=64.
// Compute: ~137 GFLOP routed through v_wmma_f32_16x16x32_bf16.
// Memory:  ~150 MB HBM traffic -> ~8us floor at 23.3 TB/s; GEMMs pipelined
//          (register-staged LDS double buffer + global_prefetch_b8), K-tiles
//          in attention moved by the Tensor Data Mover (tensor_load_to_lds).
// ---------------------------------------------------------------------------

typedef __bf16 bf16;
typedef __attribute__((ext_vector_type(16))) __bf16 v16bf;
typedef __attribute__((ext_vector_type(8)))  float  v8f;
typedef __attribute__((ext_vector_type(4)))  unsigned int v4u;
typedef __attribute__((ext_vector_type(8)))  int v8i_;
typedef __attribute__((ext_vector_type(4)))  int v4i_;

#define DEVINL __device__ __forceinline__

union Frag16 { v16bf v; uint4 q[2]; };

// A fragment (16x32 bf16, M x K): lane&15 = row M; lanes>=16 take kb=16.
// Per-lane 16 values: K = kb..kb+15  (two 16B reads).
DEVINL v16bf load_frag_a_lds(const bf16* base, int ld) {
  int lane = threadIdx.x & 31;
  int m  = lane & 15;
  int kb = (lane >> 4) << 4;            // 0 or 16
  const bf16* p = base + m * ld + kb;
  Frag16 f;
  f.q[0] = *(const uint4*)(p);          // K kb..kb+7
  f.q[1] = *(const uint4*)(p + 8);      // K kb+8..kb+15
  return f.v;
}

// B fragment (32x16 bf16, K x N) from a K-major tile: base[n*ldk + k].
// lane&15 = column N; lanes>=16 take K-half 16..31. 16 consecutive K values.
DEVINL v16bf load_frag_b_ldsT(const bf16* base, int ldk) {
  int lane = threadIdx.x & 31;
  int n  = lane & 15;
  int kb = (lane >> 4) << 4;            // 0 or 16
  const bf16* p = base + n * ldk + kb;
  Frag16 f;
  f.q[0] = *(const uint4*)(p);
  f.q[1] = *(const uint4*)(p + 8);
  return f.v;
}

// ---------------------------------------------------------------------------
// TDM: 2-D tile (tile0 x tile1 elements of 2 bytes) global -> LDS.
// D# layout per CDNA5 ISA ch.8 (group0: control/lds/global addr, group1:
// dims/strides). Issued by one wave; completion via s_wait_tensorcnt.
// This toolchain exposes the 6-arg builtin (g0, g1, g2, g3, g4, cpol);
// groups 2/3/4 are zero (2-D tile: tile_dim2/3 = 0 -> unused).
// ---------------------------------------------------------------------------
DEVINL void tdm_load_2d_bf16(unsigned lds_addr, const void* gptr,
                             unsigned tensor_dim0, unsigned tensor_dim1,
                             unsigned stride0, unsigned tile0, unsigned tile1) {
  unsigned long long ga = (unsigned long long)(size_t)gptr;
  v4u g0;
  g0[0] = 0x1u;                                        // count=1, no gather
  g0[1] = lds_addr;                                    // LDS byte address
  g0[2] = (unsigned)ga;                                // global_addr[31:0]
  g0[3] = (unsigned)((ga >> 32) & 0x1FFFFFFu)          // global_addr[56:32]
          | (2u << 30);                                // type=2 ("image")
  v8i_ g1;
  g1[0] = (int)(1u << 16);                             // data_size=1 -> 2 bytes
  g1[1] = (int)(tensor_dim0 << 16);                    // tensor_dim0[15:0]
  g1[2] = (int)((tensor_dim0 >> 16) | (tensor_dim1 << 16));
  g1[3] = (int)((tensor_dim1 >> 16) | (tile0 << 16));  // tile_dim0
  g1[4] = (int)(tile1 & 0xFFFFu);                      // tile_dim1, tile_dim2=0
  g1[5] = (int)stride0;                                // tensor_dim0_stride lo
  g1[6] = 0;
  g1[7] = 0;
  v4i_ z4 = {0, 0, 0, 0};                              // groups 2/3 unused (2-D)
  v8i_ z8 = {0, 0, 0, 0, 0, 0, 0, 0};
  __builtin_amdgcn_tensor_load_to_lds(g0, g1, z4, z4, z8, 0);
}

// ---------------------------------------------------------------------------
// fp32 -> bf16 cast (4 elems/thread; all sizes are multiples of 1024)
// ---------------------------------------------------------------------------
__global__ __launch_bounds__(256) void cvt_f32_bf16_kernel(
    const float* __restrict__ in, bf16* __restrict__ out, int n4) {
  int i = blockIdx.x * 256 + threadIdx.x;
  if (i >= n4) return;
  float4 v = *(const float4*)(in + (size_t)i * 4);
  bf16 o[4] = { (bf16)v.x, (bf16)v.y, (bf16)v.z, (bf16)v.w };
  *(uint2*)(out + (size_t)i * 4) = *(const uint2*)o;
}

// ---------------------------------------------------------------------------
// bf16 WMMA GEMM: C[M,N] = A[M,K] * B[K,N] (+bias, epilogue-specific)
// Block tile 128x128, BK=32, 256 threads = 8 waves (4x2), 32x64 per wave.
// Register-staged double buffer: global loads for step n+1 issue before the
// WMMAs of step n; global_prefetch_b8 covers step n+2.
// ---------------------------------------------------------------------------
enum { EPI_QKV = 0, EPI_BIAS_RES_F32 = 1, EPI_BIAS_RELU_BF = 2 };

template <int EPI>
__global__ __launch_bounds__(256) void gemm_bf16_kernel(
    const bf16* __restrict__ A, const bf16* __restrict__ Bm,
    const float* __restrict__ bias, const float* __restrict__ resid,
    float* __restrict__ Cf, bf16* __restrict__ Cbf,
    bf16* __restrict__ Qo, bf16* __restrict__ Ko, bf16* __restrict__ Vo,
    int M, int N, int K) {
  __shared__ __align__(16) bf16 As[128 * 32];   // row-major [m][k]
  __shared__ __align__(16) bf16 Bt[128 * 32];   // K-major   [n][k]

  const int m0 = blockIdx.y * 128, n0 = blockIdx.x * 128;
  const int tid = threadIdx.x;
  const int wid = tid >> 5, lane = tid & 31;
  const int wm = wid >> 1, wn = wid & 1;        // 4 x 2 wave grid

  // per-thread staging addresses
  const int rA = tid >> 1, cA = (tid & 1) << 4;       // A: 128 rows x 32 cols
  const int kB = tid >> 3, cB = (tid & 7) << 4;       // B: 32 rows x 128 cols
  const bf16* Ag = A + (size_t)(m0 + rA) * K + cA;
  const bf16* Bg = Bm + (size_t)kB * N + n0 + cB;

  uint4 a0 = *(const uint4*)(Ag);
  uint4 a1 = *(const uint4*)(Ag + 8);
  uint4 b0 = *(const uint4*)(Bg);
  uint4 b1 = *(const uint4*)(Bg + 8);

  v8f acc[2][4] = {};

  for (int k0 = 0; k0 < K; k0 += 32) {
    __syncthreads();                       // previous step's LDS reads done
    *(uint4*)&As[rA * 32 + cA]     = a0;
    *(uint4*)&As[rA * 32 + cA + 8] = a1;
    {
      const bf16* p0 = (const bf16*)&b0;
      const bf16* p1 = (const bf16*)&b1;
#pragma unroll
      for (int i = 0; i < 8; ++i) Bt[(cB + i) * 32 + kB] = p0[i];
#pragma unroll
      for (int i = 0; i < 8; ++i) Bt[(cB + 8 + i) * 32 + kB] = p1[i];
    }
    __syncthreads();

    if (k0 + 32 < K) {                     // stage next K-step from HBM
      const bf16* An = Ag + k0 + 32;
      const bf16* Bn = Bg + (size_t)(k0 + 32) * N;
      a0 = *(const uint4*)(An);
      a1 = *(const uint4*)(An + 8);
      b0 = *(const uint4*)(Bn);
      b1 = *(const uint4*)(Bn + 8);
      if (k0 + 64 < K) {                   // prefetch step after that
        __builtin_prefetch(Ag + k0 + 64, 0, 1);
        __builtin_prefetch(Bg + (size_t)(k0 + 64) * N, 0, 1);
      }
    }

    v16bf afr[2], bfr[4];
#pragma unroll
    for (int i = 0; i < 2; ++i)
      afr[i] = load_frag_a_lds(&As[(wm * 32 + i * 16) * 32], 32);
#pragma unroll
    for (int j = 0; j < 4; ++j)
      bfr[j] = load_frag_b_ldsT(&Bt[(wn * 64 + j * 16) * 32], 32);
#pragma unroll
    for (int i = 0; i < 2; ++i)
#pragma unroll
      for (int j = 0; j < 4; ++j)
        acc[i][j] = __builtin_amdgcn_wmma_f32_16x16x32_bf16(
            false, afr[i], false, bfr[j], (short)0, acc[i][j], false, false);
  }

  // Epilogue. C layout: vgpr r <-> M = r + 8*(lane>=16), N = lane&15.
  const int mh = lane >> 4, nl = lane & 15;
#pragma unroll
  for (int i = 0; i < 2; ++i)
#pragma unroll
    for (int j = 0; j < 4; ++j)
#pragma unroll
      for (int r = 0; r < 8; ++r) {
        int gm = m0 + wm * 32 + i * 16 + r + 8 * mh;
        int gn = n0 + wn * 64 + j * 16 + nl;
        float c = acc[i][j][r] + bias[gn];
        if constexpr (EPI == EPI_QKV) {
          // scatter into Q/K/V [B, nh, S, hd] bf16
          int which = gn >> 10;                 // 0:q 1:k 2:v
          int hh = (gn & 1023) >> 6;
          int d  = gn & 63;
          int b  = gm >> 11;
          int s  = gm & 2047;
          bf16* dst = (which == 0) ? Qo : (which == 1) ? Ko : Vo;
          dst[(((size_t)b * 16 + hh) * 2048 + s) * 64 + d] = (bf16)c;
        } else if constexpr (EPI == EPI_BIAS_RES_F32) {
          c += resid[(size_t)gm * N + gn];
          Cf[(size_t)gm * N + gn] = c;
        } else {  // EPI_BIAS_RELU_BF
          c = c > 0.f ? c : 0.f;
          Cbf[(size_t)gm * N + gn] = (bf16)c;
        }
      }
}

// ---------------------------------------------------------------------------
// Flash attention, causal. One block per (q-tile of 128, head, batch).
// 8 waves; each wave owns 16 query rows. fp32 online softmax.
// K tile moved by the Tensor Data Mover; V tile transposed manually.
// ---------------------------------------------------------------------------
__global__ __launch_bounds__(256) void attn_kernel(
    const bf16* __restrict__ Q, const bf16* __restrict__ K,
    const bf16* __restrict__ V, bf16* __restrict__ O /* [4096,1024] */) {
  __shared__ __align__(16) bf16 Ks[128 * 64];       // [key][hd]  (K-major for QK^T)
  __shared__ __align__(16) bf16 Vt[64 * 128];       // [hd][key]  (K-major for P V)
  __shared__ __align__(16) bf16 Ps[8][16 * 128];    // per-wave P tile

  const int qt = blockIdx.x, h = blockIdx.y, b = blockIdx.z;
  const int tid = threadIdx.x, wid = tid >> 5, lane = tid & 31;
  const int mh = lane >> 4, nl = lane & 15;
  const size_t base = ((size_t)b * 16 + h) * 2048 * 64;
  const bf16* Qb = Q + base;
  const bf16* Kb = K + base;
  const bf16* Vb = V + base;
  const int q0 = qt * 128;
  const float scale = 0.125f;  // 1/sqrt(64)

  // Q fragments for this wave's 16 rows (hd = 2 x K32) -- kept in registers.
  v16bf qfr[2];
  {
    int m = lane & 15, kb = (lane >> 4) << 4;
    const bf16* p = Qb + (size_t)(q0 + wid * 16 + m) * 64;
#pragma unroll
    for (int i = 0; i < 2; ++i) {
      Frag16 f;
      f.q[0] = *(const uint4*)(p + i * 32 + kb);
      f.q[1] = *(const uint4*)(p + i * 32 + kb + 8);
      qfr[i] = f.v;
    }
  }

  float rmax[8], rsum[8];
#pragma unroll
  for (int r = 0; r < 8; ++r) { rmax[r] = -1e30f; rsum[r] = 0.f; }
  v8f oacc[4] = {};

  for (int kt = 0; kt <= qt; ++kt) {
    __syncthreads();
    // K tile 128x64 via TDM (one wave issues the whole-tile DMA)
    if (wid == 0)
      tdm_load_2d_bf16((unsigned)(size_t)(const void*)&Ks[0],
                       Kb + (size_t)kt * 128 * 64,
                       /*tensor_dim0=*/64, /*tensor_dim1=*/2048,
                       /*stride0=*/64, /*tile0=*/64, /*tile1=*/128);
    {  // V tile transposed into [hd][key] (TDM cannot transpose)
      int r = tid >> 1, c = (tid & 1) << 5;
      const bf16* g = Vb + (size_t)(kt * 128 + r) * 64 + c;
#pragma unroll
      for (int i = 0; i < 4; ++i) {
        uint4 d = *(const uint4*)(g + i * 8);
        const bf16* pd = (const bf16*)&d;
#pragma unroll
        for (int e = 0; e < 8; ++e) Vt[(c + i * 8 + e) * 128 + r] = pd[e];
      }
      if (kt < qt) __builtin_prefetch(g + 128 * 64, 0, 1);   // next V tile
    }
    if (wid == 0) __builtin_amdgcn_s_wait_tensorcnt((short)0);
    __syncthreads();

    // S = Q K^T  (16 x 128 per wave), B-fragments pipelined one tile ahead
    v8f sa[8];
    v16bf bc0 = load_frag_b_ldsT(&Ks[0], 64);
    v16bf bc1 = load_frag_b_ldsT(&Ks[32], 64);
#pragma unroll
    for (int nt = 0; nt < 8; ++nt) {
      v16bf bn0 = bc0, bn1 = bc1;
      if (nt < 7) {
        bn0 = load_frag_b_ldsT(&Ks[(nt + 1) * 16 * 64], 64);
        bn1 = load_frag_b_ldsT(&Ks[(nt + 1) * 16 * 64 + 32], 64);
      }
      v8f s = {};
      s = __builtin_amdgcn_wmma_f32_16x16x32_bf16(
          false, qfr[0], false, bc0, (short)0, s, false, false);
      s = __builtin_amdgcn_wmma_f32_16x16x32_bf16(
          false, qfr[1], false, bc1, (short)0, s, false, false);
      sa[nt] = s;
      bc0 = bn0; bc1 = bn1;
    }

    // scale + causal mask + per-row max
    float tmax[8];
#pragma unroll
    for (int r = 0; r < 8; ++r) tmax[r] = -1e30f;
#pragma unroll
    for (int nt = 0; nt < 8; ++nt) {
      int kn = kt * 128 + nt * 16 + nl;
#pragma unroll
      for (int r = 0; r < 8; ++r) {
        int qm = q0 + wid * 16 + r + 8 * mh;
        float v = sa[nt][r] * scale;
        v = (kn > qm) ? -1e30f : v;
        sa[nt][r] = v;
        tmax[r] = fmaxf(tmax[r], v);
      }
    }
#pragma unroll
    for (int r = 0; r < 8; ++r) {   // reduce across the 16 lanes owning row r
      float v = tmax[r];
#pragma unroll
      for (int d = 1; d < 16; d <<= 1) v = fmaxf(v, __shfl_xor(v, d, 32));
      tmax[r] = v;
    }

    // online rescale
    float psum[8];
#pragma unroll
    for (int r = 0; r < 8; ++r) {
      float mnew = fmaxf(rmax[r], tmax[r]);
      float corr = __expf(rmax[r] - mnew);
      rmax[r] = mnew;
      rsum[r] *= corr;
#pragma unroll
      for (int t = 0; t < 4; ++t) oacc[t][r] *= corr;
      psum[r] = 0.f;
    }

    // P = exp(S - max): write per-wave LDS tile (C layout -> A layout)
    bf16* Pw = &Ps[wid][0];
#pragma unroll
    for (int nt = 0; nt < 8; ++nt)
#pragma unroll
      for (int r = 0; r < 8; ++r) {
        float p = __expf(sa[nt][r] - rmax[r]);
        psum[r] += p;
        Pw[(r + 8 * mh) * 128 + nt * 16 + nl] = (bf16)p;
      }
#pragma unroll
    for (int r = 0; r < 8; ++r) {
      float v = psum[r];
#pragma unroll
      for (int d = 1; d < 16; d <<= 1) v += __shfl_xor(v, d, 32);
      rsum[r] += v;
    }

    // O += P @ V, fragments pipelined one K-chunk ahead
    v16bf pc = load_frag_a_lds(&Pw[0], 128);
    v16bf vc[4];
#pragma unroll
    for (int t = 0; t < 4; ++t)
      vc[t] = load_frag_b_ldsT(&Vt[(t * 16) * 128], 128);
#pragma unroll
    for (int kc = 0; kc < 4; ++kc) {
      v16bf pn = pc;
      v16bf vn[4] = { vc[0], vc[1], vc[2], vc[3] };
      if (kc < 3) {
        pn = load_frag_a_lds(&Pw[(kc + 1) * 32], 128);
#pragma unroll
        for (int t = 0; t < 4; ++t)
          vn[t] = load_frag_b_ldsT(&Vt[(t * 16) * 128 + (kc + 1) * 32], 128);
      }
#pragma unroll
      for (int t = 0; t < 4; ++t)
        oacc[t] = __builtin_amdgcn_wmma_f32_16x16x32_bf16(
            false, pc, false, vc[t], (short)0, oacc[t], false, false);
      pc = pn;
#pragma unroll
      for (int t = 0; t < 4; ++t) vc[t] = vn[t];
    }
  }

  // normalize + store into [token, H] bf16 (token = b*2048 + s)
#pragma unroll
  for (int t = 0; t < 4; ++t)
#pragma unroll
    for (int r = 0; r < 8; ++r) {
      int m = wid * 16 + r + 8 * mh;
      int gm = b * 2048 + q0 + m;
      int n = t * 16 + nl;
      float o = oacc[t][r] / rsum[r];
      O[(size_t)gm * 1024 + h * 64 + n] = (bf16)o;
    }
}

// ---------------------------------------------------------------------------
// LayerNorm over H=1024, one row per block (256 threads x 4 elems).
// ---------------------------------------------------------------------------
__global__ __launch_bounds__(256) void layernorm_kernel(
    const float* __restrict__ X, const float* __restrict__ g,
    const float* __restrict__ bta, float* __restrict__ Yf,
    bf16* __restrict__ Ybf) {
  __shared__ float red[16];
  const int row = blockIdx.x;
  const int tid = threadIdx.x;
  const float* x = X + (size_t)row * 1024;
  float4 v = *(const float4*)(x + tid * 4);

  float s = v.x + v.y + v.z + v.w;
#pragma unroll
  for (int d = 1; d < 32; d <<= 1) s += __shfl_xor(s, d, 32);
  if ((tid & 31) == 0) red[tid >> 5] = s;
  __syncthreads();
  if (tid < 8) {
    float t = red[tid];
#pragma unroll
    for (int d = 1; d < 8; d <<= 1) t += __shfl_xor(t, d, 32);
    if (tid == 0) red[8] = t * (1.0f / 1024.0f);
  }
  __syncthreads();
  const float mean = red[8];

  float dx = v.x - mean, dy = v.y - mean, dz = v.z - mean, dw = v.w - mean;
  float ss = dx * dx + dy * dy + dz * dz + dw * dw;
#pragma unroll
  for (int d = 1; d < 32; d <<= 1) ss += __shfl_xor(ss, d, 32);
  if ((tid & 31) == 0) red[tid >> 5] = ss;
  __syncthreads();
  if (tid < 8) {
    float t = red[tid];
#pragma unroll
    for (int d = 1; d < 8; d <<= 1) t += __shfl_xor(t, d, 32);
    if (tid == 0) red[9] = t * (1.0f / 1024.0f);
  }
  __syncthreads();
  const float rstd = rsqrtf(red[9] + 1e-5f);

  float xs[4] = { v.x, v.y, v.z, v.w };
#pragma unroll
  for (int e = 0; e < 4; ++e) {
    int c = tid * 4 + e;
    float y = (xs[e] - mean) * rstd * g[c] + bta[c];
    if (Yf)  Yf[(size_t)row * 1024 + c] = y;
    if (Ybf) Ybf[(size_t)row * 1024 + c] = (bf16)y;
  }
}

// ---------------------------------------------------------------------------
// Host orchestration
// ---------------------------------------------------------------------------
extern "C" void kernel_launch(void* const* d_in, const int* in_sizes, int n_in,
                              void* d_out, int out_size, void* d_ws, size_t ws_size,
                              hipStream_t stream) {
  (void)in_sizes; (void)n_in; (void)out_size; (void)ws_size;
  const float* x     = (const float*)d_in[0];
  // d_in[1] attn_mask (bool) -- causal, handled analytically
  const float* qkv_w = (const float*)d_in[2];
  const float* qkv_b = (const float*)d_in[3];
  const float* out_w = (const float*)d_in[4];
  const float* out_b = (const float*)d_in[5];
  const float* w1    = (const float*)d_in[6];
  const float* b1    = (const float*)d_in[7];
  const float* w2    = (const float*)d_in[8];
  const float* b2    = (const float*)d_in[9];
  const float* ln1_g = (const float*)d_in[10];
  const float* ln1_b = (const float*)d_in[11];
  const float* ln2_g = (const float*)d_in[12];
  const float* ln2_b = (const float*)d_in[13];
  float* out = (float*)d_out;

  const size_t NT = 4096;          // B*S tokens
  const size_t H  = 1024;

  // workspace layout (with reuse: attn_bf aliases xbf, y2 aliases y1)
  char* p = (char*)d_ws;
  auto alloc_bf = [&](size_t n) { bf16* r = (bf16*)p; p += n * sizeof(bf16); return r; };
  auto alloc_f  = [&](size_t n) { float* r = (float*)p; p += n * sizeof(float); return r; };
  bf16* xbf     = alloc_bf(NT * H);       // also reused as attn_bf
  bf16* wqkv_bf = alloc_bf(H * 3 * H);
  bf16* wout_bf = alloc_bf(H * H);
  bf16* w1_bf   = alloc_bf(H * 4 * H);
  bf16* w2_bf   = alloc_bf(4 * H * H);
  bf16* qbuf    = alloc_bf(NT * H);
  bf16* kbuf    = alloc_bf(NT * H);
  bf16* vbuf    = alloc_bf(NT * H);
  bf16* xln1_bf = alloc_bf(NT * H);
  bf16* h_bf    = alloc_bf(NT * 4 * H);
  float* y1     = alloc_f(NT * H);        // also reused as y2
  float* xln1   = alloc_f(NT * H);
  bf16* attn_bf = xbf;                    // x no longer needed in bf16 after QKV
  float* y2     = y1;                     // y1 consumed by LN1 before MLP ends

  auto cvt = [&](const float* in, bf16* o, size_t n) {
    int n4 = (int)(n / 4);
    cvt_f32_bf16_kernel<<<(n4 + 255) / 256, 256, 0, stream>>>(in, o, n4);
  };
  cvt(x,     xbf,     NT * H);
  cvt(qkv_w, wqkv_bf, H * 3 * H);
  cvt(out_w, wout_bf, H * H);
  cvt(w1,    w1_bf,   H * 4 * H);
  cvt(w2,    w2_bf,   4 * H * H);

  // 1) QKV projection: [4096,1024] x [1024,3072] -> scatter Q/K/V
  gemm_bf16_kernel<EPI_QKV><<<dim3(3072 / 128, 4096 / 128), 256, 0, stream>>>(
      xbf, wqkv_bf, qkv_b, nullptr, nullptr, nullptr,
      qbuf, kbuf, vbuf, 4096, 3072, 1024);

  // 2) causal flash attention (writes attn_bf, aliasing xbf is safe: serial)
  attn_kernel<<<dim3(16, 16, 2), 256, 0, stream>>>(qbuf, kbuf, vbuf, attn_bf);

  // 3) out projection + residual -> y1 (fp32)
  gemm_bf16_kernel<EPI_BIAS_RES_F32><<<dim3(1024 / 128, 4096 / 128), 256, 0, stream>>>(
      attn_bf, wout_bf, out_b, x, y1, nullptr,
      nullptr, nullptr, nullptr, 4096, 1024, 1024);

  // 4) LN1 -> xln1 fp32 + bf16
  layernorm_kernel<<<4096, 256, 0, stream>>>(y1, ln1_g, ln1_b, xln1, xln1_bf);

  // 5) MLP up + ReLU -> h_bf
  gemm_bf16_kernel<EPI_BIAS_RELU_BF><<<dim3(4096 / 128, 4096 / 128), 256, 0, stream>>>(
      xln1_bf, w1_bf, b1, nullptr, nullptr, h_bf,
      nullptr, nullptr, nullptr, 4096, 4096, 1024);

  // 6) MLP down + residual -> y2 (fp32)
  gemm_bf16_kernel<EPI_BIAS_RES_F32><<<dim3(1024 / 128, 4096 / 128), 256, 0, stream>>>(
      h_bf, w2_bf, b2, xln1, y2, nullptr,
      nullptr, nullptr, nullptr, 4096, 1024, 4096);

  // 7) LN2 -> final fp32 output
  layernorm_kernel<<<4096, 256, 0, stream>>>(y2, ln2_g, ln2_b, out, nullptr);
}
